// DeepGraphConvModel_17300128269007
// MI455X (gfx1250) — matrix operand order repaired and checked by reading, not compiled, not run on hardware
//
#include <hip/hip_runtime.h>
#include <stdint.h>

// Problem constants (from the reference).
#define NND 50000   // nodes  (== 3125 * 16, no GEMM tail)
#define NE  800000  // edges
#define NG  64      // graphs
#define INF 64      // input feature dim
#define HID 128     // hidden dim
#define NH  4       // heads
// D_HEAD = 32

typedef __attribute__((ext_vector_type(8)))  float  v8f;
typedef __attribute__((ext_vector_type(16))) __bf16 v16bf;
typedef __attribute__((ext_vector_type(8)))  __bf16 v8bf;

// ---------------------------------------------------------------- utilities

__device__ __forceinline__ __bf16 f2bf(float f) {
  unsigned u = __float_as_uint(f);
  unsigned r = (u + 0x7FFFu + ((u >> 16) & 1u)) >> 16;   // round-to-nearest-even
  unsigned short h = (unsigned short)r;
  return __builtin_bit_cast(__bf16, h);
}

__global__ void k_f32_to_bf16(const float* __restrict__ in,
                              __bf16* __restrict__ out, int n) {
  int i = blockIdx.x * blockDim.x + threadIdx.x;
  if (i < n) out[i] = f2bf(in[i]);
}

__global__ void k_fill_f32(float* __restrict__ p, float v, int n) {
  int i = blockIdx.x * blockDim.x + threadIdx.x;
  if (i < n) p[i] = v;
}

__global__ void k_count(const int* __restrict__ gid, float* __restrict__ counts, int n) {
  int i = blockIdx.x * blockDim.x + threadIdx.x;
  if (i < n) atomicAdd(counts + gid[i], 1.0f);
}

// Monotone float atomic-max via signed-int max (positives) / unsigned min (negatives).
// Requires the cell to be initialized to -inf (0xFF800000).
__device__ __forceinline__ void atomicMaxF(float* addr, float val) {
  if (val >= 0.0f) atomicMax((int*)addr, __float_as_int(val));
  else             atomicMin((unsigned int*)addr, __float_as_uint(val));
}

// ------------------------------------------------------------- WMMA GEMM
// Out[N,128] = Xb[N,K](bf16) @ Wb[128,K](bf16)^T + bias[128], f32 accumulate.
// One wave per 16x16 output tile; block = 4 waves covering 64 columns.
// Fragment layouts used here (A and B share one contiguous k order):
//   A 16x32 bf16 : lane<16 row=lane,   K = k0+{0..15}
//                  lane>=16 row=lane-16,K = k0+{16..31}
//   B 32x16 bf16 : col = lane&15, K = k0 + (lane>>4)*16 + {0..15} (contiguous)
//   C/D 16x16 f32: VGPR r, lane t -> row = r + (t>=16 ? 8:0), col = t&15
template<int K>
__global__ __launch_bounds__(128)
void k_gemm_wmma(const __bf16* __restrict__ Xb,
                 const __bf16* __restrict__ Wb,
                 const float* __restrict__ bias,
                 float* __restrict__ Out)
{
  const int wave = threadIdx.x >> 5;
  const int lane = threadIdx.x & 31;
  const int l    = lane & 15;
  const int hi   = lane >> 4;
  const int m0   = blockIdx.x << 4;
  const int n0   = ((blockIdx.y << 2) + wave) << 4;

  const __bf16* ap = Xb + (size_t)(m0 + l) * K + hi * 16;
  const __bf16* bp = Wb + (size_t)(n0 + l) * K + hi * 16;

  const float bval = bias[n0 + l];
  v8f c = { bval, bval, bval, bval, bval, bval, bval, bval };

#pragma unroll
  for (int k0 = 0; k0 < K; k0 += 32) {
    v8bf alo = *(const v8bf*)(ap + k0);        // K = k0 + hi*16 + 0..7
    v8bf ahi = *(const v8bf*)(ap + k0 + 8);    // K = k0 + hi*16 + 8..15
    v16bf A = __builtin_shufflevector(alo, ahi,
        0,1,2,3,4,5,6,7,8,9,10,11,12,13,14,15);
    v16bf B = *(const v16bf*)(bp + k0);        // K = k0 + hi*16 + 0..15
    c = __builtin_amdgcn_wmma_f32_16x16x32_bf16(false, A, false, B,
                                                (short)0, c, false, false);
  }

  float* outp = Out + (size_t)(m0 + hi * 8) * HID + n0 + l;
#pragma unroll
  for (int r = 0; r < 8; ++r)
    outp[(size_t)r * HID] = c[r];
}

// ------------------------------------------------------------- edge phase
// One wave32 per edge; lane l covers feature elements 4l..4l+3 (head = l>>3).

__global__ __launch_bounds__(256)
void k_edge_logits(const float* __restrict__ hs, const float* __restrict__ hd,
                   const int* __restrict__ src, const int* __restrict__ dst,
                   const float* __restrict__ attn,
                   float* __restrict__ logits, float* __restrict__ mmax)
{
  int e = (blockIdx.x * blockDim.x + threadIdx.x) >> 5;
  int lane = threadIdx.x & 31;
  if (e >= NE) return;
  int s = src[e], d = dst[e];
  float4 a = ((const float4*)attn)[lane];
  float4 x = ((const float4*)(hs + (size_t)s * HID))[lane];
  float4 y = ((const float4*)(hd + (size_t)d * HID))[lane];
  float p = 0.f, v;
  v = x.x + y.x; v = v > 0.f ? v : 0.2f * v; p += v * a.x;
  v = x.y + y.y; v = v > 0.f ? v : 0.2f * v; p += v * a.y;
  v = x.z + y.z; v = v > 0.f ? v : 0.2f * v; p += v * a.z;
  v = x.w + y.w; v = v > 0.f ? v : 0.2f * v; p += v * a.w;
  // reduce the 8 lanes of each head
  p += __shfl_xor(p, 1);
  p += __shfl_xor(p, 2);
  p += __shfl_xor(p, 4);
  if ((lane & 7) == 0) {
    int h = lane >> 3;
    logits[(size_t)e * NH + h] = p;
    atomicMaxF(mmax + (size_t)d * NH + h, p);
  }
}

__global__ void k_edge_expsum(float* __restrict__ lg, const int* __restrict__ dst,
                              const float* __restrict__ mmax,
                              float* __restrict__ denom)
{
  int t = blockIdx.x * blockDim.x + threadIdx.x;
  if (t >= NE * NH) return;
  int e = t >> 2, h = t & 3;
  int d = dst[e];
  float ex = __expf(lg[t] - mmax[(size_t)d * NH + h]);
  lg[t] = ex;                                   // overwrite logits with exp
  atomicAdd(denom + (size_t)d * NH + h, ex);
}

__global__ __launch_bounds__(256)
void k_edge_alpha_agg(const float* __restrict__ hs,
                      const int* __restrict__ src, const int* __restrict__ dst,
                      const float* __restrict__ exbuf, const float* __restrict__ denom,
                      float* __restrict__ agg, float* __restrict__ alpha_out)
{
  int e = (blockIdx.x * blockDim.x + threadIdx.x) >> 5;
  int lane = threadIdx.x & 31;
  if (e >= NE) return;
  int s = src[e], d = dst[e];
  int h = lane >> 3;
  float alpha = exbuf[(size_t)e * NH + h] / denom[(size_t)d * NH + h];
  if ((lane & 7) == 0) alpha_out[(size_t)e * NH + h] = alpha;
  float4 x = ((const float4*)(hs + (size_t)s * HID))[lane];
  float* ap = agg + (size_t)d * HID + lane * 4;  // agg pre-seeded with residual
  atomicAdd(ap + 0, x.x * alpha);
  atomicAdd(ap + 1, x.y * alpha);
  atomicAdd(ap + 2, x.z * alpha);
  atomicAdd(ap + 3, x.w * alpha);
}

// ------------------------------------------------------------- BN + pool

__global__ __launch_bounds__(128)
void k_bn_stats(const float* __restrict__ x, float* __restrict__ sum,
                float* __restrict__ sumsq)
{
  int ch = threadIdx.x;
  int r0 = blockIdx.x * 64;
  int r1 = (r0 + 64 < NND) ? r0 + 64 : NND;
  float s = 0.f, s2 = 0.f;
  for (int r = r0; r < r1; ++r) {
    float v = x[(size_t)r * HID + ch];
    s += v; s2 += v * v;
  }
  atomicAdd(sum + ch, s);
  atomicAdd(sumsq + ch, s2);
}

__global__ void k_bn_final(const float* __restrict__ sum, const float* __restrict__ sumsq,
                           const float* __restrict__ gamma, const float* __restrict__ beta,
                           float* __restrict__ scale, float* __restrict__ shift)
{
  int ch = threadIdx.x;
  const float invN = 1.0f / (float)NND;
  float mean = sum[ch] * invN;
  float var  = sumsq[ch] * invN - mean * mean;
  float sc   = gamma[ch] * rsqrtf(var + 1e-5f);
  scale[ch] = sc;
  shift[ch] = beta[ch] - mean * sc;
}

// y = relu(x*scale+shift), stored in place (becomes next layer's input),
// and pooled per graph. graph_ids are sorted -> run-length-compress atomics.
__global__ __launch_bounds__(128)
void k_bn_apply_pool(float* __restrict__ x, const float* __restrict__ scale,
                     const float* __restrict__ shift, const int* __restrict__ gid,
                     float* __restrict__ pooled)
{
  int ch = threadIdx.x;
  float sc = scale[ch], sh = shift[ch];
  int r0 = blockIdx.x * 64;
  int r1 = (r0 + 64 < NND) ? r0 + 64 : NND;
  if (r0 >= r1) return;
  int cur = gid[r0];
  float acc = 0.f;
  for (int r = r0; r < r1; ++r) {
    int g = gid[r];
    float v = fmaxf(x[(size_t)r * HID + ch] * sc + sh, 0.f);
    x[(size_t)r * HID + ch] = v;
    if (g != cur) { atomicAdd(pooled + (size_t)cur * HID + ch, acc); acc = 0.f; cur = g; }
    acc += v;
  }
  atomicAdd(pooled + (size_t)cur * HID + ch, acc);
}

__global__ void k_pool_final(const float* __restrict__ pooled,
                             const float* __restrict__ counts,
                             float* __restrict__ feats)
{
  int g = blockIdx.x, ch = threadIdx.x;
  float c = fmaxf(counts[g], 1.0f);
  feats[(size_t)g * HID + ch] = pooled[(size_t)g * HID + ch] / c;
}

// ------------------------------------------------------------- launch

extern "C" void kernel_launch(void* const* d_in, const int* in_sizes, int n_in,
                              void* d_out, int out_size, void* d_ws, size_t ws_size,
                              hipStream_t stream)
{
  (void)in_sizes; (void)n_in; (void)out_size; (void)ws_size;

  // jax pytree order: top-level dict insertion order is
  // {features, params, src, dst, graph_ids}; param dicts flatten sorted by key.
  const float* features = (const float*)d_in[0];
  const float* attn0    = (const float*)d_in[1];
  const float* b_dst0   = (const float*)d_in[2];
  const float* b_res0   = (const float*)d_in[3];
  const float* b_src0   = (const float*)d_in[4];
  const float* bn_beta0 = (const float*)d_in[5];
  const float* bn_gam0  = (const float*)d_in[6];
  const float* w_dst0   = (const float*)d_in[7];
  const float* w_res0   = (const float*)d_in[8];
  const float* w_src0   = (const float*)d_in[9];
  const float* attn1    = (const float*)d_in[10];
  const float* b_dst1   = (const float*)d_in[11];
  const float* b_src1   = (const float*)d_in[12];
  const float* bn_beta1 = (const float*)d_in[13];
  const float* bn_gam1  = (const float*)d_in[14];
  const float* w_dst1   = (const float*)d_in[15];
  const float* w_src1   = (const float*)d_in[16];
  const int*   src      = (const int*)d_in[17];
  const int*   dst      = (const int*)d_in[18];
  const int*   gid      = (const int*)d_in[19];

  // Output: feats0[64,128], feats1[64,128], attn0[800000,4], attn1[800000,4]
  float* out_feat0 = (float*)d_out;
  float* out_feat1 = out_feat0 + (size_t)NG * HID;
  float* out_attn0 = out_feat1 + (size_t)NG * HID;
  float* out_attn1 = out_attn0 + (size_t)NE * NH;

  // Workspace carve-out (~130 MB).
  char* ws = (char*)d_ws;
  size_t off = 0;
  auto take = [&](size_t bytes) -> void* {
    void* p = ws + off;
    off = (off + bytes + 255) & ~(size_t)255;
    return p;
  };
  float*  hs     = (float*)take((size_t)NND * HID * 4);
  float*  hd     = (float*)take((size_t)NND * HID * 4);
  float*  bufA   = (float*)take((size_t)NND * HID * 4);  // layer0 agg -> layer1 x
  float*  bufB   = (float*)take((size_t)NND * HID * 4);  // layer1 agg
  __bf16* xb     = (__bf16*)take((size_t)NND * HID * 2);
  __bf16* wb0    = (__bf16*)take((size_t)HID * HID * 2);
  __bf16* wb1    = (__bf16*)take((size_t)HID * HID * 2);
  __bf16* wb2    = (__bf16*)take((size_t)HID * HID * 2);
  float*  mmax   = (float*)take((size_t)NND * NH * 4);
  float*  denom  = (float*)take((size_t)NND * NH * 4);
  float*  lg     = (float*)take((size_t)NE * NH * 4);
  float*  sums   = (float*)take(HID * 4);
  float*  sumsq  = (float*)take(HID * 4);
  float*  scale  = (float*)take(HID * 4);
  float*  shift  = (float*)take(HID * 4);
  float*  pooled = (float*)take((size_t)NG * HID * 4);
  float*  counts = (float*)take(NG * 4);

  const float NEG_INF = -__builtin_huge_valf();
  const dim3 gemm_grid(NND / 16, HID / 64);   // (3125, 2), 4 waves/block

  // ---- per-graph node counts (used by both layers)
  k_fill_f32<<<1, 64, 0, stream>>>(counts, 0.f, NG);
  k_count<<<(NND + 255) / 256, 256, 0, stream>>>(gid, counts, NND);

  // =============================== Layer 0 (K = 64)
  k_f32_to_bf16<<<(NND * INF + 255) / 256, 256, 0, stream>>>(features, xb, NND * INF);
  k_f32_to_bf16<<<(HID * INF + 255) / 256, 256, 0, stream>>>(w_src0, wb0, HID * INF);
  k_f32_to_bf16<<<(HID * INF + 255) / 256, 256, 0, stream>>>(w_dst0, wb1, HID * INF);
  k_f32_to_bf16<<<(HID * INF + 255) / 256, 256, 0, stream>>>(w_res0, wb2, HID * INF);

  k_gemm_wmma<INF><<<gemm_grid, 128, 0, stream>>>(xb, wb0, b_src0, hs);
  k_gemm_wmma<INF><<<gemm_grid, 128, 0, stream>>>(xb, wb1, b_dst0, hd);
  k_gemm_wmma<INF><<<gemm_grid, 128, 0, stream>>>(xb, wb2, b_res0, bufA); // residual seeds agg

  k_fill_f32<<<(NND * NH + 255) / 256, 256, 0, stream>>>(mmax, NEG_INF, NND * NH);
  k_fill_f32<<<(NND * NH + 255) / 256, 256, 0, stream>>>(denom, 0.f, NND * NH);

  k_edge_logits<<<NE / 8, 256, 0, stream>>>(hs, hd, src, dst, attn0, lg, mmax);
  k_edge_expsum<<<(NE * NH + 255) / 256, 256, 0, stream>>>(lg, dst, mmax, denom);
  k_edge_alpha_agg<<<NE / 8, 256, 0, stream>>>(hs, src, dst, lg, denom, bufA, out_attn0);

  k_fill_f32<<<1, 128, 0, stream>>>(sums, 0.f, HID);
  k_fill_f32<<<1, 128, 0, stream>>>(sumsq, 0.f, HID);
  k_fill_f32<<<(NG * HID + 255) / 256, 256, 0, stream>>>(pooled, 0.f, NG * HID);

  k_bn_stats<<<(NND + 63) / 64, 128, 0, stream>>>(bufA, sums, sumsq);
  k_bn_final<<<1, 128, 0, stream>>>(sums, sumsq, bn_gam0, bn_beta0, scale, shift);
  k_bn_apply_pool<<<(NND + 63) / 64, 128, 0, stream>>>(bufA, scale, shift, gid, pooled);
  k_pool_final<<<NG, 128, 0, stream>>>(pooled, counts, out_feat0);

  // =============================== Layer 1 (K = 128, identity residual)
  k_f32_to_bf16<<<(NND * HID + 255) / 256, 256, 0, stream>>>(bufA, xb, NND * HID);
  k_f32_to_bf16<<<(HID * HID + 255) / 256, 256, 0, stream>>>(w_src1, wb0, HID * HID);
  k_f32_to_bf16<<<(HID * HID + 255) / 256, 256, 0, stream>>>(w_dst1, wb1, HID * HID);

  k_gemm_wmma<HID><<<gemm_grid, 128, 0, stream>>>(xb, wb0, b_src1, hs);
  k_gemm_wmma<HID><<<gemm_grid, 128, 0, stream>>>(xb, wb1, b_dst1, hd);

  // identity residual seeds agg
  hipMemcpyAsync(bufB, bufA, (size_t)NND * HID * 4, hipMemcpyDeviceToDevice, stream);

  k_fill_f32<<<(NND * NH + 255) / 256, 256, 0, stream>>>(mmax, NEG_INF, NND * NH);
  k_fill_f32<<<(NND * NH + 255) / 256, 256, 0, stream>>>(denom, 0.f, NND * NH);

  k_edge_logits<<<NE / 8, 256, 0, stream>>>(hs, hd, src, dst, attn1, lg, mmax);
  k_edge_expsum<<<(NE * NH + 255) / 256, 256, 0, stream>>>(lg, dst, mmax, denom);
  k_edge_alpha_agg<<<NE / 8, 256, 0, stream>>>(hs, src, dst, lg, denom, bufB, out_attn1);

  k_fill_f32<<<1, 128, 0, stream>>>(sums, 0.f, HID);
  k_fill_f32<<<1, 128, 0, stream>>>(sumsq, 0.f, HID);
  k_fill_f32<<<(NG * HID + 255) / 256, 256, 0, stream>>>(pooled, 0.f, NG * HID);

  k_bn_stats<<<(NND + 63) / 64, 128, 0, stream>>>(bufB, sums, sumsq);
  k_bn_final<<<1, 128, 0, stream>>>(sums, sumsq, bn_gam1, bn_beta1, scale, shift);
  k_bn_apply_pool<<<(NND + 63) / 64, 128, 0, stream>>>(bufB, scale, shift, gid, pooled);
  k_pool_final<<<NG, 128, 0, stream>>>(pooled, counts, out_feat1);
}
